// HeteroGNNModel_81475529605803
// MI455X (gfx1250) — matrix operand order repaired and checked by reading, not compiled, not run on hardware
//
#include <hip/hip_runtime.h>
#include <hip/hip_bf16.h>

typedef __attribute__((ext_vector_type(16))) __bf16 v16bf;
typedef __attribute__((ext_vector_type(8)))  __bf16 v8bf;
typedef __attribute__((ext_vector_type(2)))  __bf16 v2bf;
typedef __attribute__((ext_vector_type(8)))  float  v8f;

#define NH 100000
#define NC 80000
#define NO 20000
#define EE 200000
#define DIN 64
#define DHID 512
#define DOUT 256

#define ASTR 40   // A LDS stride (ushorts): 80B, 16B-aligned rows, conflict-free frag reads
#define BSTR 34   // B LDS stride (ushorts): 68B, odd dword multiplier -> conflict-free stores

__device__ __forceinline__ unsigned short f2bf(float f) {
  unsigned u = __float_as_uint(f);
  unsigned r = u + 0x7FFFu + ((u >> 16) & 1u);   // round-to-nearest-even
  return (unsigned short)(r >> 16);
}
__device__ __forceinline__ unsigned pk2bf(float a, float b) {
  return (unsigned)f2bf(a) | ((unsigned)f2bf(b) << 16);
}

// C[M,N] = maybe_relu( A[M,K] @ W[K,N] + (bias? bias : 0) + (beta? C : 0) )
// 256 threads = 8 waves; block tile 64(M) x 64(N); wave tile 16x32 (2 WMMAs, shared A frag).
__global__ __launch_bounds__(256)
void gemm_bf16_wmma(const float* __restrict__ A, const float* __restrict__ W,
                    const float* __restrict__ bias, float* __restrict__ C,
                    int M, int N, int K, int beta, int relu) {
  __shared__ alignas(16) unsigned short As[64 * ASTR];   // [m][k] bf16, padded
  __shared__ alignas(16) unsigned short Bs[64 * BSTR];   // [n][k] bf16 (transposed), padded

  const int tid  = threadIdx.x;
  const int lane = tid & 31;
  const int wave = tid >> 5;
  const int wm = wave >> 1;            // 0..3
  const int wn = wave & 1;             // 0..1
  const int m0 = blockIdx.x * 64;
  const int n0 = blockIdx.y * 64;

  v8f acc0 = {}, acc1 = {};

  // Fragment coordinates (CDNA5 ISA 7.12.2)
  const int am  = wm * 16 + (lane & 15);   // A row in tile
  const int ah  = lane >> 4;               // A k-half: k-base = ah*16
  const int bn0 = wn * 32 + (lane & 15);   // first B col
  const int bn1 = bn0 + 16;                // second B col
  const int kb  = (lane >> 4) * 16;        // B k-base (ushorts)

  for (int kt = 0; kt < K; kt += 32) {
    // ---- Stage A: 64x32 fp32 -> bf16. 512 float4-quads, 2 per thread.
#pragma unroll
    for (int i = 0; i < 2; ++i) {
      int q  = tid + i * 256;
      int m  = q >> 3;
      int kq = (q & 7) << 2;
      float4 v = make_float4(0.f, 0.f, 0.f, 0.f);
      int gm = m0 + m;
      if (gm < M) v = *(const float4*)(A + (size_t)gm * K + kt + kq);
      uint2 pp;
      pp.x = pk2bf(v.x, v.y);
      pp.y = pk2bf(v.z, v.w);
      *(uint2*)(As + m * ASTR + kq) = pp;          // ds_store_b64, 8B-aligned
    }
    // ---- Stage B transposed: W[K,N] -> Bs[n][k]. 64n x 16 k-pairs, 4 per thread.
    //      Two coalesced b32 global reads per unit; one b32 LDS store (k,k+1 packed).
#pragma unroll
    for (int i = 0; i < 4; ++i) {
      int q  = tid + i * 256;
      int n  = q & 63;
      int kp = (q >> 6) << 1;                      // 0,2,...,30
      const float* wp = W + (size_t)(kt + kp) * N + n0 + n;
      unsigned p = pk2bf(wp[0], wp[N]);
      *(unsigned*)(Bs + n * BSTR + kp) = p;        // ds_store_b32, conflict-free (17n mod 64)
    }
    // Prefetch next K tile (global_prefetch_b8) while this tile is consumed.
    if (kt + 32 < K) {
      int m  = tid >> 3;
      int gm = m0 + m; if (gm >= M) gm = M - 1;
      __builtin_prefetch(A + (size_t)gm * K + kt + 32 + ((tid & 7) << 2), 0, 0);
      __builtin_prefetch(W + (size_t)(kt + 32 + ((tid >> 6) << 1)) * N + n0 + (tid & 63), 0, 0);
    }
    __syncthreads();

    // ---- A fragment: two 16B LDS loads (row am: k = ah*16..+7 and ah*16+8..+15)
    const v8bf* arow = (const v8bf*)(As + am * ASTR);
    v8bf alo = arow[2 * ah];
    v8bf ahi = arow[2 * ah + 1];
    v16bf a;
#pragma unroll
    for (int i = 0; i < 8; ++i) { a[i] = alo[i]; a[i + 8] = ahi[i]; }

    // ---- B fragments: 8x b32 each (col bn: k = kb..kb+15), 4B-aligned
    const v2bf* bp0 = (const v2bf*)(Bs + bn0 * BSTR + kb);
    const v2bf* bp1 = (const v2bf*)(Bs + bn1 * BSTR + kb);
    v16bf b0, b1;
#pragma unroll
    for (int j = 0; j < 8; ++j) {
      v2bf x0 = bp0[j], x1 = bp1[j];
      b0[2 * j] = x0[0]; b0[2 * j + 1] = x0[1];
      b1[2 * j] = x1[0]; b1[2 * j + 1] = x1[1];
    }

    acc0 = __builtin_amdgcn_wmma_f32_16x16x32_bf16(false, a, false, b0, (short)0, acc0, false, false);
    acc1 = __builtin_amdgcn_wmma_f32_16x16x32_bf16(false, a, false, b1, (short)0, acc1, false, false);
    __syncthreads();
  }

  // ---- Epilogue. C/D layout: lane -> N = lane&15 ; VGPR j -> M = j + 8*(lane>=16)
  const int ngl = n0 + wn * 32 + (lane & 15);
#pragma unroll
  for (int j = 0; j < 8; ++j) {
    int mg = m0 + wm * 16 + j + ((lane >> 4) << 3);
    if (mg < M) {
      size_t base = (size_t)mg * N;
      float v0 = acc0[j], v1 = acc1[j];
      if (bias) { v0 += bias[ngl];        v1 += bias[ngl + 16]; }
      if (beta) { v0 += C[base + ngl];    v1 += C[base + ngl + 16]; }
      if (relu) { v0 = fmaxf(v0, 0.f);    v1 = fmaxf(v1, 0.f); }
      C[base + ngl]      = v0;
      C[base + ngl + 16] = v1;
    }
  }
}

// agg[dst[e]] += hsrc[src[e]]  (256-wide rows, float4 chunks, 64 chunks/edge)
__global__ __launch_bounds__(256)
void scatter_add_kernel(const float* __restrict__ hsrc, const int* __restrict__ src,
                        const int* __restrict__ dst, float* __restrict__ agg, int ne) {
  long long tid = (long long)blockIdx.x * blockDim.x + threadIdx.x;
  int e = (int)(tid >> 6);
  if (e >= ne) return;
  int c = ((int)tid & 63) << 2;
  int s = src[e], d = dst[e];
  const float4 v = *(const float4*)(hsrc + (size_t)s * DOUT + c);
  float* p = agg + (size_t)d * DOUT + c;
  atomicAdd(p + 0, v.x); atomicAdd(p + 1, v.y);
  atomicAdd(p + 2, v.z); atomicAdd(p + 3, v.w);
}

__global__ __launch_bounds__(256)
void zero_kernel(float* __restrict__ p, long long n) {
  long long i = (long long)blockIdx.x * blockDim.x + threadIdx.x;
  long long stride = (long long)gridDim.x * blockDim.x;
  for (; i < n; i += stride) p[i] = 0.f;
}

// wsum[L,d] = sum over {r=d, d+3, d+6} of Wroot[L,r]; bsum likewise from brel.
__global__ __launch_bounds__(256)
void sum_root_kernel(const float* __restrict__ Wroot, const float* __restrict__ brel,
                     float* __restrict__ wsum, float* __restrict__ bsum) {
  const int PER = DOUT * DOUT;
  int idx = blockIdx.x * blockDim.x + threadIdx.x;
  if (idx < 2 * 3 * PER) {
    int L = idx / (3 * PER);
    int rem = idx - L * 3 * PER;
    int d = rem / PER;
    int e = rem - d * PER;
    const float* base = Wroot + (size_t)L * 9 * PER;
    wsum[idx] = base[(size_t)d * PER + e] + base[(size_t)(d + 3) * PER + e] +
                base[(size_t)(d + 6) * PER + e];
  }
  if (idx < 2 * 3 * DOUT) {
    int L = idx / (3 * DOUT);
    int rem = idx - L * 3 * DOUT;
    int d = rem / DOUT;
    int e = rem - d * DOUT;
    const float* bb = brel + (size_t)L * 9 * DOUT;
    bsum[idx] = bb[(size_t)d * DOUT + e] + bb[(size_t)(d + 3) * DOUT + e] +
                bb[(size_t)(d + 6) * DOUT + e];
  }
}

// out[row] = dot(h[row,:256], w) + b   (one wave32 per row, 8 rows/block)
__global__ __launch_bounds__(256)
void head_dot_kernel(const float* __restrict__ h, const float* __restrict__ w,
                     const float* __restrict__ b, float* __restrict__ out, int nrows) {
  int row = blockIdx.x * 8 + (threadIdx.x >> 5);
  int lane = threadIdx.x & 31;
  if (row >= nrows) return;
  const float* hp = h + (size_t)row * DOUT;
  float s = 0.f;
#pragma unroll
  for (int i = 0; i < DOUT / 32; ++i) s += hp[lane + i * 32] * w[lane + i * 32];
#pragma unroll
  for (int off2 = 16; off2 > 0; off2 >>= 1) s += __shfl_down(s, off2, 32);
  if (lane == 0) out[row] = s + b[0];
}

extern "C" void kernel_launch(void* const* d_in, const int* in_sizes, int n_in,
                              void* d_out, int out_size, void* d_ws, size_t ws_size,
                              hipStream_t stream) {
  const float* xH = (const float*)d_in[0];
  const float* xC = (const float*)d_in[1];
  const float* xO = (const float*)d_in[2];
  const float* encW1 = (const float*)d_in[3];
  const float* encb1 = (const float*)d_in[4];
  const float* encW2 = (const float*)d_in[5];
  const float* encb2 = (const float*)d_in[6];
  const float* convWrel  = (const float*)d_in[7];
  const float* convbrel  = (const float*)d_in[8];
  const float* convWroot = (const float*)d_in[9];
  const float* headWH = (const float*)d_in[10];
  const float* headbH = (const float*)d_in[11];
  const float* headWC = (const float*)d_in[12];
  const float* headbC = (const float*)d_in[13];
  const int* ei[9];
  for (int r = 0; r < 9; ++r) ei[r] = (const int*)d_in[14 + r];

  // Workspace layout (floats): h_a | h_b (also encoder hidden scratch) | agg | wsum | bsum
  float* ws = (float*)d_ws;
  const long long TOT = (long long)(NH + NC + NO);
  float* h_a  = ws;
  float* h_b  = h_a + TOT * DOUT;          // 51.2M floats (>= NH*DHID for hid scratch)
  float* agg  = h_b + TOT * DOUT;          // NH*DOUT max
  float* wsum = agg + (long long)NH * DOUT;
  float* bsum = wsum + 2LL * 3 * DOUT * DOUT;

  const int Ns[3] = {NH, NC, NO};
  const long long off[3] = {0, (long long)NH * DOUT, (long long)(NH + NC) * DOUT};
  const float* xs[3] = {xH, xC, xO};

  dim3 blk(256);

  // Pre-sum root weights/biases per (layer, dst-type): 3x fewer root GEMMs.
  sum_root_kernel<<<dim3((2 * 3 * DOUT * DOUT + 255) / 256), blk, 0, stream>>>(
      convWroot, convbrel, wsum, bsum);

  // Encoders: h_a[t] = relu(relu(x @ W1 + b1) @ W2 + b2), hidden staged in h_b.
  for (int t = 0; t < 3; ++t) {
    int M = Ns[t];
    gemm_bf16_wmma<<<dim3((M + 63) / 64, DHID / 64), blk, 0, stream>>>(
        xs[t], encW1 + (size_t)t * DIN * DHID, encb1 + (size_t)t * DHID,
        h_b, M, DHID, DIN, /*beta=*/0, /*relu=*/1);
    gemm_bf16_wmma<<<dim3((M + 63) / 64, DOUT / 64), blk, 0, stream>>>(
        h_b, encW2 + (size_t)t * DHID * DOUT, encb2 + (size_t)t * DOUT,
        h_a + off[t], M, DOUT, DHID, /*beta=*/0, /*relu=*/1);
  }

  // 2 conv layers, ping-pong h_a <-> h_b.
  float* hcur = h_a;
  float* hnext = h_b;
  for (int L = 0; L < 2; ++L) {
    for (int d = 0; d < 3; ++d) {
      int M = Ns[d];
      // Root term seeds the accumulator: hnext[d] = hcur[d] @ wsum[L,d] + bsum[L,d]
      gemm_bf16_wmma<<<dim3((M + 63) / 64, DOUT / 64), blk, 0, stream>>>(
          hcur + off[d], wsum + (size_t)(L * 3 + d) * DOUT * DOUT,
          bsum + (size_t)(L * 3 + d) * DOUT,
          hnext + off[d], M, DOUT, DOUT, /*beta=*/0, /*relu=*/0);
      // Relations with dst=d are r = d, d+3, d+6 with src types 0,1,2.
      for (int k = 0; k < 3; ++k) {
        int r = 3 * k + d;
        int s = k;
        zero_kernel<<<dim3(2048), blk, 0, stream>>>(agg, (long long)M * DOUT);
        scatter_add_kernel<<<dim3((EE * 64 + 255) / 256), blk, 0, stream>>>(
            hcur + off[s], ei[r], ei[r] + EE, agg, EE);
        gemm_bf16_wmma<<<dim3((M + 63) / 64, DOUT / 64), blk, 0, stream>>>(
            agg, convWrel + (size_t)(L * 9 + r) * DOUT * DOUT, nullptr,
            hnext + off[d], M, DOUT, DOUT, /*beta=*/1, /*relu=*/(k == 2) ? 1 : 0);
      }
    }
    float* t = hcur; hcur = hnext; hnext = t;
  }
  // After 2 swaps, final features are back in h_a (== hcur).

  float* out = (float*)d_out;
  head_dot_kernel<<<dim3((NH + 7) / 8), blk, 0, stream>>>(hcur + off[0], headWH, headbH, out, NH);
  head_dot_kernel<<<dim3((NC + 7) / 8), blk, 0, stream>>>(hcur + off[1], headWC, headbC, out + NH, NC);
}
